// LowRankTransformerModel_36627481101166
// MI455X (gfx1250) — matrix operand order repaired and checked by reading, not compiled, not run on hardware
//
#include <hip/hip_runtime.h>
#include <hip/hip_bf16.h>

// ---------------------------------------------------------------------------
// LowRankTransformerModel forward for MI455X (gfx1250, wave32, WMMA + TDM).
// Weight-bandwidth-bound transformer: f16 WMMA GEMMs (v_wmma_f32_16x16x32_f16,
// fp32 accum), weights streamed into LDS by the Tensor Data Mover
// (tensor_load_to_lds, double-buffered, s_wait_tensorcnt), fp32 residual
// stream, structural attention masks computed inline.
// ---------------------------------------------------------------------------

typedef __attribute__((ext_vector_type(16))) _Float16     v16h;
typedef __attribute__((ext_vector_type(8)))  _Float16     v8h;
typedef __attribute__((ext_vector_type(8)))  float        v8f;
typedef __attribute__((ext_vector_type(4)))  unsigned int v4u;
typedef __attribute__((ext_vector_type(8)))  int          v8i;
typedef __attribute__((ext_vector_type(4)))  int          v4i;

constexpr int kB  = 64;
constexpr int kS  = 83;
constexpr int kD  = 1024;
constexpr int kL  = 12;
constexpr int kH  = 16;
constexpr int kHD = 64;
constexpr int kND = 20;
constexpr int kBS = kB * kS;        // 5312 = 332 * 16
constexpr int kD3 = 3 * kD;         // 3072
constexpr int kD4 = 4 * kD;         // 4096
constexpr int kMT = kBS / 16;       // 332 M-tiles
constexpr int kKC = 32;             // K-chunk per TDM stage (one WMMA k-step)

enum { EPI_QKV = 0, EPI_RES = 1, EPI_GELU = 2 };

#if defined(__has_builtin)
#if __has_builtin(__builtin_amdgcn_tensor_load_to_lds)
#define ATHENA_HAVE_TDM 1
#endif
#if __has_builtin(__builtin_amdgcn_s_wait_tensorcnt)
#define WAIT_TENSORCNT(n) __builtin_amdgcn_s_wait_tensorcnt(n)
#else
#define WAIT_TENSORCNT(n) asm volatile("s_wait_tensorcnt %0" ::"i"(n))
#endif
#else
#define WAIT_TENSORCNT(n)
#endif

// ---------------------------------------------------------------------------
// Reductions (wave32 shuffles + cross-wave LDS)
// ---------------------------------------------------------------------------
__device__ inline float block_reduce_sum(float v, float* sh) {
  #pragma unroll
  for (int off = 16; off > 0; off >>= 1) v += __shfl_xor(v, off, 32);
  const int wave = threadIdx.x >> 5, lane = threadIdx.x & 31;
  if (lane == 0) sh[wave] = v;
  __syncthreads();
  float r = (threadIdx.x < 8) ? sh[threadIdx.x] : 0.0f;
  if (wave == 0) {
    #pragma unroll
    for (int off = 4; off > 0; off >>= 1) r += __shfl_xor(r, off, 32);
    if (lane == 0) sh[0] = r;
  }
  __syncthreads();
  r = sh[0];
  __syncthreads();
  return r;
}

// ---------------------------------------------------------------------------
// Embedding: h = xs @ W_in + b_in + pos   (K = 20, VALU is fine here)
// ---------------------------------------------------------------------------
__global__ __launch_bounds__(256) void embed_kernel(
    const float* __restrict__ xs, const float* __restrict__ W_in,
    const float* __restrict__ b_in, const float* __restrict__ pos_shared,
    const float* __restrict__ pos_last3, float* __restrict__ h) {
  __shared__ float xrow[kND];
  const int tok = blockIdx.x;
  const int s   = tok % kS;
  if (threadIdx.x < kND) xrow[threadIdx.x] = xs[(size_t)tok * kND + threadIdx.x];
  __syncthreads();
  #pragma unroll
  for (int i = 0; i < kD / 256; ++i) {
    const int d = threadIdx.x + i * 256;
    float acc = b_in[d];
    acc += (s < 80) ? pos_shared[(s & 3) * kD + d] : pos_last3[(s - 80) * kD + d];
    #pragma unroll
    for (int j = 0; j < kND; ++j) acc += xrow[j] * W_in[j * kD + d];
    h[(size_t)tok * kD + d] = acc;
  }
}

// ---------------------------------------------------------------------------
// LayerNorm -> f16 activation (GEMM A operand)
// ---------------------------------------------------------------------------
__global__ __launch_bounds__(256) void ln_to_f16_kernel(
    const float* __restrict__ x, const float* __restrict__ g,
    const float* __restrict__ b, _Float16* __restrict__ out) {
  __shared__ float sh[8];
  const size_t row = blockIdx.x;
  const float* xr = x + row * kD;
  float vals[kD / 256];
  float s = 0.0f, s2 = 0.0f;
  #pragma unroll
  for (int i = 0; i < kD / 256; ++i) {
    const float t = xr[threadIdx.x + i * 256];
    vals[i] = t; s += t; s2 += t * t;
  }
  const float mean = block_reduce_sum(s, sh) * (1.0f / kD);
  const float var  = block_reduce_sum(s2, sh) * (1.0f / kD) - mean * mean;
  const float rs   = rsqrtf(var + 1e-5f);
  #pragma unroll
  for (int i = 0; i < kD / 256; ++i) {
    const int d = threadIdx.x + i * 256;
    out[row * kD + d] = (_Float16)((vals[i] - mean) * rs * g[d] + b[d]);
  }
}

// ---------------------------------------------------------------------------
// Weight transpose + fp32 -> f16:  w[K,N] -> wT[N,K] (contiguous K per column)
// ---------------------------------------------------------------------------
__global__ __launch_bounds__(256) void transpose_to_f16_kernel(
    const float* __restrict__ w, _Float16* __restrict__ wT, int K, int N) {
  __shared__ float tile[32][33];
  const int n0 = blockIdx.x * 32, k0 = blockIdx.y * 32;
  const int tx = threadIdx.x & 31, ty = threadIdx.x >> 5;
  #pragma unroll
  for (int i = 0; i < 32; i += 8)
    tile[ty + i][tx] = w[(size_t)(k0 + ty + i) * N + (n0 + tx)];
  __syncthreads();
  #pragma unroll
  for (int i = 0; i < 32; i += 8)
    wT[(size_t)(n0 + ty + i) * K + (k0 + tx)] = (_Float16)tile[tx][ty + i];
}

__device__ inline float gelu_tanh(float x) {
  const float c = 0.7978845608028654f;  // sqrt(2/pi)
  return 0.5f * x * (1.0f + tanhf(c * (x + 0.044715f * x * x * x)));
}

// ---------------------------------------------------------------------------
// Tensor Data Mover: 2-D f16 tile load Global -> LDS.
// D# packing per CDNA5 ISA 8.3/8.4:
//  g0: [1:0] count=1 | [63:32] lds_addr | [120:64] global_addr | [127:126] type=2
//  g1: [17:16] data_size=1 (2B) | [79:48] tensor_dim0 | [111:80] tensor_dim1 |
//      [127:112] tile_dim0 | [143:128] tile_dim1 | [207:160] tensor_dim0_stride
//  groups 2/3: zero (2-D tensor), workgroup_mask=0 (not in a cluster).
// ---------------------------------------------------------------------------
#if defined(ATHENA_HAVE_TDM)
__device__ inline void tdm_load_2d_f16(const void* lds_dst,
                                       const _Float16* __restrict__ gsrc,
                                       unsigned tile_x, unsigned tile_y,
                                       unsigned long long tdim_x,
                                       unsigned long long tdim_y,
                                       unsigned long long stride_x) {
  const unsigned long long ga = (unsigned long long)gsrc;
  const unsigned lds32 = (unsigned)(unsigned long long)lds_dst;  // LDS aperture low bits
  v4u g0;
  g0[0] = 1u;                                          // count=1, user descriptor
  g0[1] = lds32;                                       // lds_addr
  g0[2] = (unsigned)(ga & 0xffffffffull);              // global_addr[31:0]
  g0[3] = (unsigned)((ga >> 32) & 0x01ffffffull) | (2u << 30);  // addr hi | type=2

  const unsigned long long u0 = (1ull << 16)                       // data_size = 2B
                              | ((tdim_x & 0xffffull) << 48);      // tensor_dim0 lo16
  const unsigned long long u1 = ((tdim_x >> 16) & 0xffffull)       // tensor_dim0 hi16
                              | ((tdim_y & 0xffffffffull) << 16)   // tensor_dim1
                              | ((unsigned long long)(tile_x & 0xffffu) << 48);
  const unsigned long long u2 = (unsigned long long)(tile_y & 0xffffu)  // tile_dim1
                              | ((stride_x & 0xffffffffull) << 32);     // stride lo32
  const unsigned long long u3 = (stride_x >> 32) & 0xffffull;           // stride hi16
  v8i g1;
  g1[0] = (int)u0; g1[1] = (int)(u0 >> 32);
  g1[2] = (int)u1; g1[3] = (int)(u1 >> 32);
  g1[4] = (int)u2; g1[5] = (int)(u2 >> 32);
  g1[6] = (int)u3; g1[7] = (int)(u3 >> 32);
  const v4i z4 = {0, 0, 0, 0};
#if defined(__clang_major__) && (__clang_major__ >= 23)
  const v8i z8 = {0, 0, 0, 0, 0, 0, 0, 0};
  __builtin_amdgcn_tensor_load_to_lds(g0, g1, z4, z4, z8, 0);
#else
  __builtin_amdgcn_tensor_load_to_lds(g0, g1, z4, z4, 0);
#endif
}
#endif

// ---------------------------------------------------------------------------
// One WMMA k-step (K=32) from LDS tiles, contiguous k order, same for A and B.
// A 16x32: M = lane&15, K = (lane>=16 ? 16 : 0) + {0..15}
// B 32x16: N = lane&15, K = (lane>=16 ? 16 : 0) + {0..15}
// ---------------------------------------------------------------------------
__device__ inline void wmma_lds_step(const _Float16* As, const _Float16* Bs,
                                     int lane, int wave, v8f acc[4]) {
  const int ak = (lane >> 4) << 4;
  const v8h alo = *(const v8h*)(As + (lane & 15) * kKC + ak);
  const v8h ahi = *(const v8h*)(As + (lane & 15) * kKC + ak + 8);
  v16h a;
  #pragma unroll
  for (int i = 0; i < 8; ++i) { a[i] = alo[i]; a[i + 8] = ahi[i]; }
  const int bk = (lane >> 4) << 4;
  #pragma unroll
  for (int t = 0; t < 4; ++t) {
    const v16h b =
        *(const v16h*)(Bs + (wave * 64 + t * 16 + (lane & 15)) * kKC + bk);
    acc[t] = __builtin_amdgcn_wmma_f32_16x16x32_f16(
        false, a, false, b, (short)0, acc[t], false, false);
  }
}

// ---------------------------------------------------------------------------
// Fused GEMM:  out = epi(A[f16, M x K] @ Bt^T[f16, N x K] + bias)
// Block = 128 threads = 4 waves -> 16 rows x 256 cols; A tile shared via LDS.
// TDM streams A (16x32) and B (256x32) tiles into double-buffered LDS;
// wave 0 issues descriptors and gates with s_wait_tensorcnt.
// ---------------------------------------------------------------------------
template <int EPI>
__global__ __launch_bounds__(128) void gemm_f16_wmma_kernel(
    const _Float16* __restrict__ A, const _Float16* __restrict__ Bt,
    const float* __restrict__ bias, int N, int K,
    float* __restrict__ resid,        // EPI_RES: fp32 residual, updated in place
    _Float16* __restrict__ out16,     // EPI_GELU: f16 [M,N] output
    _Float16* __restrict__ qbuf, _Float16* __restrict__ kbuf,
    _Float16* __restrict__ vbuf) {    // EPI_QKV: [B,H,S,HD] f16 scatter
  __shared__ alignas(64) _Float16 Asm[2][16 * kKC];    // 2 x 1 KB
  __shared__ alignas(64) _Float16 Bsm[2][256 * kKC];   // 2 x 16 KB
  const int lane  = threadIdx.x & 31;
  const int wave  = threadIdx.x >> 5;
  const int mtile = blockIdx.x;
  const int nblk  = blockIdx.y * 256;
  const int NC    = K / kKC;

  v8f acc[4] = {v8f{}, v8f{}, v8f{}, v8f{}};

#if defined(ATHENA_HAVE_TDM)
  if (wave == 0) {
    tdm_load_2d_f16(Asm[0], A + (size_t)mtile * 16 * K, kKC, 16,
                    (unsigned long long)K, (unsigned long long)kBS,
                    (unsigned long long)K);
    tdm_load_2d_f16(Bsm[0], Bt + (size_t)nblk * K, kKC, 256,
                    (unsigned long long)K, (unsigned long long)N,
                    (unsigned long long)K);
  }
  for (int c = 0; c < NC; ++c) {
    if (wave == 0) {
      if (c + 1 < NC) {  // prefetch next chunk into the other buffer
        const int k0 = (c + 1) * kKC;
        const int nb = (c + 1) & 1;
        tdm_load_2d_f16(Asm[nb], A + (size_t)mtile * 16 * K + k0, kKC, 16,
                        (unsigned long long)K, (unsigned long long)kBS,
                        (unsigned long long)K);
        tdm_load_2d_f16(Bsm[nb], Bt + (size_t)nblk * K + k0, kKC, 256,
                        (unsigned long long)K, (unsigned long long)N,
                        (unsigned long long)K);
        WAIT_TENSORCNT(2);   // chunk c's two TDM ops complete (in-order)
      } else {
        WAIT_TENSORCNT(0);
      }
    }
    __syncthreads();                       // buffer c&1 ready for all waves
    wmma_lds_step(Asm[c & 1], Bsm[c & 1], lane, wave, acc);
    __syncthreads();                       // done reading before reuse
  }
#else
  // Fallback: cooperative vector-load LDS fill (no TDM builtin available)
  for (int c = 0; c < NC; ++c) {
    const int k0 = c * kKC;
    for (int r = threadIdx.x; r < 256; r += 128) {
      const v16h* src = (const v16h*)(Bt + (size_t)(nblk + r) * K + k0);
      v16h* dst = (v16h*)(Bsm[0] + r * kKC);
      dst[0] = src[0]; dst[1] = src[1];
    }
    if (threadIdx.x < 16) {
      const v16h* src = (const v16h*)(A + (size_t)(mtile * 16 + threadIdx.x) * K + k0);
      v16h* dst = (v16h*)(Asm[0] + threadIdx.x * kKC);
      dst[0] = src[0]; dst[1] = src[1];
    }
    __syncthreads();
    wmma_lds_step(Asm[0], Bsm[0], lane, wave, acc);
    __syncthreads();
  }
#endif

  // C/D layout: element (M = mtile*16 + r + 8*(lane>=16), N = nblk + wave*64 + t*16 + (lane&15))
  const int row_hi = (lane >> 4) << 3;
  #pragma unroll
  for (int t = 0; t < 4; ++t) {
    const int n = nblk + wave * 64 + t * 16 + (lane & 15);
    const float bv = bias[n];
    #pragma unroll
    for (int r = 0; r < 8; ++r) {
      const int m = mtile * 16 + row_hi + r;
      const float val = acc[t][r] + bv;
      if (EPI == EPI_RES) {
        float* p = resid + (size_t)m * N + n;
        *p = *p + val;
      } else if (EPI == EPI_GELU) {
        out16[(size_t)m * N + n] = (_Float16)gelu_tanh(val);
      } else {  // EPI_QKV: n -> {q,k,v}, head, dim; m -> (batch, seq)
        const int part = n >> 10;
        const int w    = n & 1023;
        const int head = w >> 6;
        const int hd   = w & 63;
        const int bi   = m / kS;
        const int si   = m - bi * kS;
        _Float16* dst  = (part == 0) ? qbuf : ((part == 1) ? kbuf : vbuf);
        dst[(((size_t)(bi * kH + head)) * kS + si) * kHD + hd] = (_Float16)val;
      }
    }
  }
}

// ---------------------------------------------------------------------------
// Structural attention mask (replaces the reference's additive -1e9 masks).
// Layers 0-1: causal within blocks of 4 tokens (rows 0..79); rows 80..82 self.
// Layers >=2: active set = {every 4th token} U {80,81,82}. Active queries see
// only active keys; inactive queries see plain causal attention (the uniform
// -1e9 row offset cancels inside softmax).
// ---------------------------------------------------------------------------
__device__ inline bool attn_allowed(int layer, int qi, int ki) {
  if (ki > qi) return false;  // causal
  if (layer < 2) {
    if (qi >= 80) return ki == qi;
    return (ki < 80) && ((ki >> 2) == (qi >> 2));
  }
  const bool q_active = (qi >= 80) || ((qi & 3) == 3);
  if (!q_active) return true;
  return (ki >= 80) || ((ki & 3) == 3);
}

// One wave per query row; lanes cover keys {lane, lane+32, lane+64}.
__global__ __launch_bounds__(256) void attention_kernel(
    const _Float16* __restrict__ q, const _Float16* __restrict__ k,
    const _Float16* __restrict__ v, _Float16* __restrict__ out, int layer) {
  __shared__ float shq[8][kHD];
  __shared__ float shp[8][96];
  const int bh   = blockIdx.x;
  const int bi   = bh >> 4;
  const int hi   = bh & 15;
  const int wave = threadIdx.x >> 5;
  const int lane = threadIdx.x & 31;
  const int qi   = blockIdx.y * 8 + wave;
  const bool active = qi < kS;
  const size_t head_base = ((size_t)bh) * kS * kHD;

  if (active) {
    const _Float16* qr = q + head_base + (size_t)qi * kHD;
    shq[wave][lane]      = (float)qr[lane];
    shq[wave][lane + 32] = (float)qr[lane + 32];
  }
  __syncthreads();

  float lg[3] = {-3.0e38f, -3.0e38f, -3.0e38f};
  if (active) {
    #pragma unroll
    for (int j = 0; j < 3; ++j) {
      const int ki = lane + j * 32;
      if (ki < kS && attn_allowed(layer, qi, ki)) {
        const _Float16* kr = k + head_base + (size_t)ki * kHD;
        float s = 0.0f;
        #pragma unroll
        for (int d = 0; d < kHD; ++d) s += shq[wave][d] * (float)kr[d];
        lg[j] = s * 0.125f;  // 1/sqrt(64)
      }
    }
  }
  float mx = fmaxf(fmaxf(lg[0], lg[1]), lg[2]);
  #pragma unroll
  for (int off = 16; off > 0; off >>= 1) mx = fmaxf(mx, __shfl_xor(mx, off, 32));
  float p[3], se = 0.0f;
  #pragma unroll
  for (int j = 0; j < 3; ++j) {
    p[j] = (lg[j] > -1.0e37f) ? __expf(lg[j] - mx) : 0.0f;
    se += p[j];
  }
  #pragma unroll
  for (int off = 16; off > 0; off >>= 1) se += __shfl_xor(se, off, 32);
  const float inv = 1.0f / se;
  if (active) {
    shp[wave][lane]      = p[0] * inv;
    shp[wave][lane + 32] = p[1] * inv;
    shp[wave][lane + 64] = p[2] * inv;
  }
  __syncthreads();

  if (active) {
    const _Float16* vb = v + head_base;
    float o0 = 0.0f, o1 = 0.0f;
    for (int kk = 0; kk < kS; ++kk) {
      const float pw = shp[wave][kk];
      o0 += pw * (float)vb[(size_t)kk * kHD + lane];
      o1 += pw * (float)vb[(size_t)kk * kHD + lane + 32];
    }
    _Float16* orow = out + ((size_t)(bi * kS + qi)) * kD + hi * kHD;
    orow[lane]      = (_Float16)o0;
    orow[lane + 32] = (_Float16)o1;
  }
}

// ---------------------------------------------------------------------------
// Final LN + head:  out[token] = ln(h) . W_out + b_out
// ---------------------------------------------------------------------------
__global__ __launch_bounds__(256) void final_out_kernel(
    const float* __restrict__ x, const float* __restrict__ g,
    const float* __restrict__ b, const float* __restrict__ wout,
    const float* __restrict__ bout, float* __restrict__ out) {
  __shared__ float sh[8];
  const size_t row = blockIdx.x;
  const float* xr = x + row * kD;
  float vals[kD / 256];
  float s = 0.0f, s2 = 0.0f;
  #pragma unroll
  for (int i = 0; i < kD / 256; ++i) {
    const float t = xr[threadIdx.x + i * 256];
    vals[i] = t; s += t; s2 += t * t;
  }
  const float mean = block_reduce_sum(s, sh) * (1.0f / kD);
  const float var  = block_reduce_sum(s2, sh) * (1.0f / kD) - mean * mean;
  const float rs   = rsqrtf(var + 1e-5f);
  float dot = 0.0f;
  #pragma unroll
  for (int i = 0; i < kD / 256; ++i) {
    const int d = threadIdx.x + i * 256;
    dot += ((vals[i] - mean) * rs * g[d] + b[d]) * wout[d];
  }
  dot = block_reduce_sum(dot, sh);
  if (threadIdx.x == 0) out[row] = dot + bout[0];
}

// ---------------------------------------------------------------------------
// Host-side orchestration (all launches on `stream`, graph-capture safe)
// ---------------------------------------------------------------------------
static inline size_t align_up(size_t x) { return (x + 255) & ~(size_t)255; }

extern "C" void kernel_launch(void* const* d_in, const int* in_sizes, int n_in,
                              void* d_out, int out_size, void* d_ws, size_t ws_size,
                              hipStream_t stream) {
  (void)in_sizes; (void)n_in; (void)out_size; (void)ws_size;
  const float* xs         = (const float*)d_in[0];
  /* ys = d_in[1] unused */
  const float* pos_shared = (const float*)d_in[2];
  const float* pos_last3  = (const float*)d_in[3];
  const float* W_in       = (const float*)d_in[4];
  const float* b_in       = (const float*)d_in[5];
  const float* ln1_g      = (const float*)d_in[6];
  const float* ln1_b      = (const float*)d_in[7];
  const float* Wqkv       = (const float*)d_in[8];
  const float* bqkv       = (const float*)d_in[9];
  const float* Wproj      = (const float*)d_in[10];
  const float* bproj      = (const float*)d_in[11];
  const float* ln2_g      = (const float*)d_in[12];
  const float* ln2_b      = (const float*)d_in[13];
  const float* Wfc        = (const float*)d_in[14];
  const float* bfc        = (const float*)d_in[15];
  const float* Wmp        = (const float*)d_in[16];
  const float* bmp        = (const float*)d_in[17];
  const float* lnf_g      = (const float*)d_in[18];
  const float* lnf_b      = (const float*)d_in[19];
  const float* W_out      = (const float*)d_in[20];
  const float* b_out      = (const float*)d_in[21];

  // Workspace carve-up (~128 MB total)
  char* ws = (char*)d_ws;
  size_t off = 0;
  float*     h      = (float*)(ws + off);      off += align_up((size_t)kBS * kD * 4);
  _Float16*  a16    = (_Float16*)(ws + off);   off += align_up((size_t)kBS * kD * 2);
  _Float16*  q16    = (_Float16*)(ws + off);   off += align_up((size_t)kBS * kD * 2);
  _Float16*  k16    = (_Float16*)(ws + off);   off += align_up((size_t)kBS * kD * 2);
  _Float16*  v16    = (_Float16*)(ws + off);   off += align_up((size_t)kBS * kD * 2);
  _Float16*  attn16 = (_Float16*)(ws + off);   off += align_up((size_t)kBS * kD * 2);
  _Float16*  fc16   = (_Float16*)(ws + off);   off += align_up((size_t)kBS * kD4 * 2);
  _Float16*  wT     = (_Float16*)(ws + off);   off += align_up((size_t)kD4 * kD * 2);

  embed_kernel<<<kBS, 256, 0, stream>>>(xs, W_in, b_in, pos_shared, pos_last3, h);

  for (int l = 0; l < kL; ++l) {
    // --- attention block ---
    ln_to_f16_kernel<<<kBS, 256, 0, stream>>>(h, ln1_g + (size_t)l * kD,
                                              ln1_b + (size_t)l * kD, a16);
    transpose_to_f16_kernel<<<dim3(kD3 / 32, kD / 32), 256, 0, stream>>>(
        Wqkv + (size_t)l * kD * kD3, wT, kD, kD3);
    gemm_f16_wmma_kernel<EPI_QKV><<<dim3(kMT, kD3 / 256), 128, 0, stream>>>(
        a16, wT, bqkv + (size_t)l * kD3, kD3, kD, nullptr, nullptr, q16, k16, v16);
    attention_kernel<<<dim3(kB * kH, (kS + 7) / 8), 256, 0, stream>>>(
        q16, k16, v16, attn16, l);
    transpose_to_f16_kernel<<<dim3(kD / 32, kD / 32), 256, 0, stream>>>(
        Wproj + (size_t)l * kD * kD, wT, kD, kD);
    gemm_f16_wmma_kernel<EPI_RES><<<dim3(kMT, kD / 256), 128, 0, stream>>>(
        attn16, wT, bproj + (size_t)l * kD, kD, kD, h, nullptr, nullptr, nullptr, nullptr);

    // --- MLP block ---
    ln_to_f16_kernel<<<kBS, 256, 0, stream>>>(h, ln2_g + (size_t)l * kD,
                                              ln2_b + (size_t)l * kD, a16);
    transpose_to_f16_kernel<<<dim3(kD4 / 32, kD / 32), 256, 0, stream>>>(
        Wfc + (size_t)l * kD * kD4, wT, kD, kD4);
    gemm_f16_wmma_kernel<EPI_GELU><<<dim3(kMT, kD4 / 256), 128, 0, stream>>>(
        a16, wT, bfc + (size_t)l * kD4, kD4, kD, nullptr, fc16, nullptr, nullptr, nullptr);
    transpose_to_f16_kernel<<<dim3(kD / 32, kD4 / 32), 256, 0, stream>>>(
        Wmp + (size_t)l * kD4 * kD, wT, kD4, kD);
    gemm_f16_wmma_kernel<EPI_RES><<<dim3(kMT, kD / 256), 128, 0, stream>>>(
        fc16, wT, bmp + (size_t)l * kD, kD, kD4, h, nullptr, nullptr, nullptr, nullptr);
  }

  final_out_kernel<<<kBS, 256, 0, stream>>>(h, lnf_g, lnf_b, W_out, b_out,
                                            (float*)d_out);
}
